// Head_71004399337648
// MI455X (gfx1250) — matrix operand order repaired and checked by reading, not compiled, not run on hardware
//
#include <hip/hip_runtime.h>
#include <hip/hip_bf16.h>
#include <stdint.h>

// Problem constants (match reference)
#define B_ 8
#define T_ 2048
#define C_ 1024
#define H_ 128

typedef __attribute__((ext_vector_type(16))) __bf16 bf16x16;
typedef __attribute__((ext_vector_type(8)))  float  v8f;
typedef __attribute__((ext_vector_type(4)))  unsigned int v4u;
typedef __attribute__((ext_vector_type(8)))  int v8i;
typedef __attribute__((ext_vector_type(4)))  int v4i;

union BF16F {
    bf16x16 v;
    uint4 q[2];
    unsigned short u[16];
};

union U16x8 {
    unsigned short s[8];
    uint4 q;
};

// native f32 -> bf16 (hardware v_cvt, round-to-nearest-even)
__device__ __forceinline__ unsigned short bfbits(float x) {
    union { __bf16 b; unsigned short s; } u;
    u.b = (__bf16)x;
    return u.s;
}

__device__ __forceinline__ v8f wmma_bf16(bf16x16 a, bf16x16 b, v8f c) {
    // (neg_a, A, neg_b, B, c_mod, C, reuse_a, reuse_b)
    return __builtin_amdgcn_wmma_f32_16x16x32_bf16(false, a, false, b, (short)0, c, false, false);
}

__device__ __forceinline__ int RF(unsigned int x) {
    return __builtin_amdgcn_readfirstlane((int)x);
}

// ---------------------------------------------------------------------------
// TDM: async 2-D tile load Global -> LDS (CDNA5 ISA sec.8 descriptor).
// data_size = 4 bytes; dims/strides in 4-byte units. Groups 2/3 zero (2-D).
// This toolchain's builtin is the 6-arg form:
//   (uint32x4 g0, int32x8 g1, int32x4, int32x4, int32x8, i32 cpol)
// ---------------------------------------------------------------------------
__device__ __forceinline__ void tdm_load_2d(uint64_t gaddr, unsigned int ldsoff,
                                            int td0, int td1, int tile0, int tile1,
                                            int stride0) {
    v4u g0;
    g0.x = 1u;                                   // count=1, user descriptor
    g0.y = (unsigned int)RF(ldsoff);             // lds_addr (bytes)
    g0.z = (unsigned int)RF((unsigned int)gaddr);            // global_addr[31:0]
    g0.w = (unsigned int)RF((unsigned int)(gaddr >> 32)) | (2u << 30);  // [56:32] | type=2
    v8i g1;
    g1[0] = 2 << 16;                             // wg_mask=0, data_size=2 (4B)
    g1[1] = (td0 & 0xFFFF) << 16;                // tensor_dim0[15:0]
    g1[2] = ((td0 >> 16) & 0xFFFF) | ((td1 & 0xFFFF) << 16);
    g1[3] = ((td1 >> 16) & 0xFFFF) | ((tile0 & 0xFFFF) << 16);
    g1[4] = (tile1 & 0xFFFF);                    // tile_dim1 ; tile_dim2 = 0
    g1[5] = stride0;                             // tensor_dim0_stride[31:0]
    g1[6] = 0;
    g1[7] = 0;
    v4i z4 = {0, 0, 0, 0};
    v8i z8 = {0, 0, 0, 0, 0, 0, 0, 0};
    __builtin_amdgcn_tensor_load_to_lds(g0, g1, z4, z4, z8, 0);
}

__device__ __forceinline__ unsigned int lds_off(const void* p) {
    return (unsigned int)(uintptr_t)p;  // generic LDS pointer: low 32 bits = LDS offset
}

// ---------------------------------------------------------------------------
// Kernel 0: transpose + convert weights fp32 [C,H] -> bf16 WT [3][H][C]
// order: 0 = w_query, 1 = w_key, 2 = w_value
// ---------------------------------------------------------------------------
__global__ void prep_wt(const float* __restrict__ wq,
                        const float* __restrict__ wk,
                        const float* __restrict__ wv,
                        unsigned short* __restrict__ wt) {
    int idx = blockIdx.x * blockDim.x + threadIdx.x;
    const int total = 3 * H_ * C_;
    if (idx >= total) return;
    int which = idx / (H_ * C_);
    int r = idx % (H_ * C_);
    int h = r / C_;
    int c = r % C_;
    const float* src = (which == 0) ? wq : (which == 1) ? wk : wv;
    wt[idx] = bfbits(src[(size_t)c * H_ + h]);
}

// ---------------------------------------------------------------------------
// Kernel 1: fused QKV projection (context read ONCE).
// Grid: B*T/16 blocks of 128 threads (4 waves). Each wave: 16 tokens x 32 heads
// for all three projections (6 WMMAs per 32-wide K step). Q,K row-major bf16;
// V transposed to VT[b][h][t] via per-wave LDS 16x16 transpose.
// ---------------------------------------------------------------------------
__global__ void __launch_bounds__(128)
proj_kernel(const float* __restrict__ ctx,
            const unsigned short* __restrict__ wt,   // [3][H][C] bf16
            unsigned short* __restrict__ Qbf,
            unsigned short* __restrict__ Kbf,
            unsigned short* __restrict__ VT) {
    __shared__ __align__(16) unsigned short vlds[4][16][16];

    const int tid  = threadIdx.x;
    const int wave = tid >> 5;
    const int lane = tid & 31;
    const int lm   = lane & 15;
    const int hi   = lane >> 4;          // 0 or 1 (half-wave)

    const int tile  = blockIdx.x;
    const int b     = tile / (T_ / 16);
    const int tbase = (tile % (T_ / 16)) * 16;

    v8f acc[3][2];
#pragma unroll
    for (int w = 0; w < 3; ++w)
#pragma unroll
        for (int j = 0; j < 2; ++j)
            acc[w][j] = (v8f){0.f, 0.f, 0.f, 0.f, 0.f, 0.f, 0.f, 0.f};

    const float* arow = ctx + ((size_t)b * T_ + tbase + lm) * C_;

#pragma unroll 4
    for (int kk = 0; kk < C_ / 32; ++kk) {
        // A fragment: 16x32 bf16 (lanes0-15: K0-15; lanes16-31: K16-31)
        BF16F a;
        const float* p0 = arow + kk * 32 + hi * 16;
        const float* p1 = arow + kk * 32 + hi * 16 + 8;
        float4 f0 = *(const float4*)(p0);
        float4 f1 = *(const float4*)(p0 + 4);
        float4 f2 = *(const float4*)(p1);
        float4 f3 = *(const float4*)(p1 + 4);
        a.v[0] = (__bf16)f0.x; a.v[1] = (__bf16)f0.y; a.v[2] = (__bf16)f0.z; a.v[3] = (__bf16)f0.w;
        a.v[4] = (__bf16)f1.x; a.v[5] = (__bf16)f1.y; a.v[6] = (__bf16)f1.z; a.v[7] = (__bf16)f1.w;
        a.v[8]  = (__bf16)f2.x; a.v[9]  = (__bf16)f2.y; a.v[10] = (__bf16)f2.z; a.v[11] = (__bf16)f2.w;
        a.v[12] = (__bf16)f3.x; a.v[13] = (__bf16)f3.y; a.v[14] = (__bf16)f3.z; a.v[15] = (__bf16)f3.w;

#pragma unroll
        for (int w = 0; w < 3; ++w) {
#pragma unroll
            for (int j = 0; j < 2; ++j) {
                const int h = (wave * 2 + j) * 16 + lm;
                bf16x16 bw = *(const bf16x16*)(wt + ((size_t)w * H_ + h) * C_ + kk * 32 + hi * 16);
                acc[w][j] = wmma_bf16(a.v, bw, acc[w][j]);
            }
        }
    }

#pragma unroll
    for (int j = 0; j < 2; ++j) {
        const int hb = (wave * 2 + j) * 16;
#pragma unroll
        for (int v = 0; v < 8; ++v) {
            const int row = tbase + v + hi * 8;
            const size_t o = ((size_t)b * T_ + row) * H_ + hb + lm;
            Qbf[o] = bfbits(acc[0][j][v]);
            Kbf[o] = bfbits(acc[1][j][v]);
            vlds[wave][v + hi * 8][lm] = bfbits(acc[2][j][v]);
        }
        U16x8 pk;
#pragma unroll
        for (int r2 = 0; r2 < 8; ++r2)
            pk.s[r2] = vlds[wave][hi * 8 + r2][lm];
        uint4* dst = (uint4*)(VT + ((size_t)b * H_ + hb + lm) * T_ + tbase + hi * 8);
        *dst = pk.q;
    }
}

// ---------------------------------------------------------------------------
// Kernel 2: causal flash attention, TDM-fed.
// Grid: (T/64, B), 128 threads (4 waves). The workgroup owns a 64-query
// supertile; each wave computes 16 query rows. K/V 32-key tiles are streamed
// into double-buffered LDS by the Tensor Data Mover (wave 0 issues, waits on
// TENSORcnt, barrier publishes), overlapping DMA with WMMA.
// ---------------------------------------------------------------------------
__global__ void __launch_bounds__(128)
attn_kernel(const unsigned short* __restrict__ Q,   // [b][t][h] bf16
            const unsigned short* __restrict__ K,   // [b][t][h] bf16
            const unsigned short* __restrict__ VT,  // [b][h][t] bf16
            float* __restrict__ out) {
    __shared__ __align__(16) unsigned short kls[2][32][128];   // 16 KB
    __shared__ __align__(16) unsigned short vls[2][128][32];   // 16 KB
    __shared__ __align__(16) unsigned short plds[4][16][32];   //  4 KB

    const int tid  = threadIdx.x;
    const int wave = tid >> 5;
    const int lane = tid & 31;
    const int lm   = lane & 15;
    const int hi   = lane >> 4;

    const int b    = blockIdx.y;
    const int base = blockIdx.x * 64;
    const int q0   = base + wave * 16;
    const float scale = 0.03125f;  // 1/sqrt(C), C=1024

    // uniform causal bound for the whole workgroup: keys < base+64
    const int nblocks = base / 32 + 2;

    // resident Q A-fragments: 16 x 128 bf16
    BF16F aq[4];
#pragma unroll
    for (int kk = 0; kk < 4; ++kk) {
        const unsigned short* qb = Q + ((size_t)b * T_ + q0 + lm) * H_ + kk * 32;
        aq[kk].q[0] = *(const uint4*)(qb + hi * 16);
        aq[kk].q[1] = *(const uint4*)(qb + hi * 16 + 8);
    }

    v8f o[8];
#pragma unroll
    for (int t = 0; t < 8; ++t) o[t] = (v8f){0.f, 0.f, 0.f, 0.f, 0.f, 0.f, 0.f, 0.f};
    float mrow[8], lrow[8];
#pragma unroll
    for (int v = 0; v < 8; ++v) { mrow[v] = -INFINITY; lrow[v] = 0.f; }

    // --- TDM issue helper: K tile 32 keys x 128 h (rows=32, rowlen 64 x 4B),
    //     V tile 128 h x 32 t (rows=128, rowlen 16 x 4B)
    auto issue_kv = [&](int kb) {
        const int jb = kb * 32;
        const int buf = kb & 1;
        uint64_t kga = (uint64_t)(uintptr_t)(K + ((size_t)b * T_ + jb) * H_);
        tdm_load_2d(kga, lds_off(&kls[buf][0][0]),
                    /*td0=*/H_ / 2, /*td1=*/B_ * T_, /*tile0=*/H_ / 2, /*tile1=*/32,
                    /*stride0=*/H_ / 2);
        uint64_t vga = (uint64_t)(uintptr_t)(VT + (size_t)b * H_ * T_ + jb);
        tdm_load_2d(vga, lds_off(&vls[buf][0][0]),
                    /*td0=*/T_ / 2, /*td1=*/B_ * H_, /*tile0=*/16, /*tile1=*/H_,
                    /*stride0=*/T_ / 2);
    };

    if (wave == 0) issue_kv(0);

    for (int kb = 0; kb < nblocks; ++kb) {
        const int jb  = kb * 32;
        const int buf = kb & 1;

        if (wave == 0) {
            if (kb + 1 < nblocks) {
                issue_kv(kb + 1);                         // prefetch next block
                __builtin_amdgcn_s_wait_tensorcnt(2);     // current pair complete
            } else {
                __builtin_amdgcn_s_wait_tensorcnt(0);
            }
        }
        __syncthreads();   // publish K/V tiles for this block

        // ---- S = Q * K^T for 32 keys (two 16x16 tiles), K from LDS
        v8f s0 = (v8f){0.f, 0.f, 0.f, 0.f, 0.f, 0.f, 0.f, 0.f};
        v8f s1 = s0;
#pragma unroll
        for (int kk = 0; kk < 4; ++kk) {
            bf16x16 bk0 = *(const bf16x16*)&kls[buf][lm][kk * 32 + hi * 16];
            bf16x16 bk1 = *(const bf16x16*)&kls[buf][16 + lm][kk * 32 + hi * 16];
            s0 = wmma_bf16(aq[kk].v, bk0, s0);
            s1 = wmma_bf16(aq[kk].v, bk1, s1);
        }

        // ---- scale + causal mask (also handles fully-masked blocks above diag)
        const int rg = q0 + hi * 8;
        const int c0 = jb + lm;
        const bool need_mask = (jb + 31 > q0);
#pragma unroll
        for (int v = 0; v < 8; ++v) {
            float x0 = s0[v] * scale;
            float x1 = s1[v] * scale;
            if (need_mask) {
                const int r = rg + v;
                if (c0 > r)      x0 = -INFINITY;
                if (c0 + 16 > r) x1 = -INFINITY;
            }
            s0[v] = x0;
            s1[v] = x1;
        }

        // ---- online softmax: row stats via 16-lane butterflies (wave32 halves)
        float al[8];
#pragma unroll
        for (int v = 0; v < 8; ++v) {
            float x = fmaxf(s0[v], s1[v]);
            x = fmaxf(x, __shfl_xor(x, 1, 16));
            x = fmaxf(x, __shfl_xor(x, 2, 16));
            x = fmaxf(x, __shfl_xor(x, 4, 16));
            x = fmaxf(x, __shfl_xor(x, 8, 16));
            const float mnew = fmaxf(mrow[v], x);
            const float p0 = __expf(s0[v] - mnew);
            const float p1 = __expf(s1[v] - mnew);
            s0[v] = p0;
            s1[v] = p1;
            float sum = p0 + p1;
            sum += __shfl_xor(sum, 1, 16);
            sum += __shfl_xor(sum, 2, 16);
            sum += __shfl_xor(sum, 4, 16);
            sum += __shfl_xor(sum, 8, 16);
            const float a = __expf(mrow[v] - mnew);
            lrow[v] = lrow[v] * a + sum;
            mrow[v] = mnew;
            al[v] = a;
        }

#pragma unroll
        for (int t = 0; t < 8; ++t)
#pragma unroll
            for (int v = 0; v < 8; ++v)
                o[t][v] *= al[v];

        // ---- re-layout P (C/D layout) -> A fragment 16x32 via per-wave LDS
#pragma unroll
        for (int v = 0; v < 8; ++v) {
            const int r = v + hi * 8;
            plds[wave][r][lm]      = bfbits(s0[v]);
            plds[wave][r][16 + lm] = bfbits(s1[v]);
        }
        BF16F pa;
        pa.q[0] = *(const uint4*)&plds[wave][lm][hi * 16];
        pa.q[1] = *(const uint4*)&plds[wave][lm][hi * 16 + 8];

        // ---- O += P * V, V from LDS (columns contiguous along t)
#pragma unroll
        for (int t = 0; t < 8; ++t) {
            bf16x16 bv = *(const bf16x16*)&vls[buf][t * 16 + lm][hi * 16];
            o[t] = wmma_bf16(pa.v, bv, o[t]);
        }

        __syncthreads();   // all waves done reading buf before TDM reuses it
    }

    // ---- epilogue: O / l, write fp32 [b][t][h]
    float rl[8];
#pragma unroll
    for (int v = 0; v < 8; ++v) rl[v] = 1.0f / lrow[v];
#pragma unroll
    for (int t = 0; t < 8; ++t) {
#pragma unroll
        for (int v = 0; v < 8; ++v) {
            const int r = q0 + v + hi * 8;
            out[((size_t)b * T_ + r) * H_ + t * 16 + lm] = o[t][v] * rl[v];
        }
    }
}

// ---------------------------------------------------------------------------
// Host launcher
// ---------------------------------------------------------------------------
extern "C" void kernel_launch(void* const* d_in, const int* in_sizes, int n_in,
                              void* d_out, int out_size, void* d_ws, size_t ws_size,
                              hipStream_t stream) {
    (void)in_sizes; (void)n_in; (void)out_size; (void)ws_size;
    const float* ctx = (const float*)d_in[0];
    const float* wk  = (const float*)d_in[1];
    const float* wq  = (const float*)d_in[2];
    const float* wv  = (const float*)d_in[3];
    float* out = (float*)d_out;

    // workspace layout (bf16 = ushort)
    unsigned short* Qbf = (unsigned short*)d_ws;                 // B*T*H
    unsigned short* Kbf = Qbf + (size_t)B_ * T_ * H_;            // B*T*H
    unsigned short* VT  = Kbf + (size_t)B_ * T_ * H_;            // B*H*T
    unsigned short* WT  = VT  + (size_t)B_ * T_ * H_;            // 3*H*C

    const int wt_total = 3 * H_ * C_;
    prep_wt<<<(wt_total + 255) / 256, 256, 0, stream>>>(wq, wk, wv, WT);

    proj_kernel<<<dim3(B_ * T_ / 16), 128, 0, stream>>>(ctx, WT, Qbf, Kbf, VT);

    attn_kernel<<<dim3(T_ / 64, B_), 128, 0, stream>>>(Qbf, Kbf, VT, out);
}
